// GCN_80633716015250
// MI455X (gfx1250) — compile-verified, matrix operand order checked by reading, not run on hardware
//
#include <hip/hip_runtime.h>

// ---------- CDNA5 WMMA types ----------
typedef __attribute__((ext_vector_type(16))) __bf16 v16bf;
typedef __attribute__((ext_vector_type(8)))  __bf16 v8bf;
typedef __attribute__((ext_vector_type(8)))  float  v8f;

// float -> bf16 with round-to-nearest-even (bit-level, no reliance on cvt codegen)
__device__ __forceinline__ __bf16 f2bf(float f) {
  unsigned int u = __builtin_bit_cast(unsigned int, f);
  u += 0x7FFFu + ((u >> 16) & 1u);
  unsigned short s = (unsigned short)(u >> 16);
  return __builtin_bit_cast(__bf16, s);
}

// ---------- utility kernels ----------
__global__ void fill_zero_kernel(float* __restrict__ p, int n) {
  int i = blockIdx.x * blockDim.x + threadIdx.x;
  if (i < n) p[i] = 0.0f;
}

__global__ void degree_kernel(const int* __restrict__ src, const int* __restrict__ dst,
                              float* __restrict__ degOut, float* __restrict__ degIn, int nE) {
  int e = blockIdx.x * blockDim.x + threadIdx.x;
  if (e < nE) {
    atomicAdd(&degOut[src[e]], 1.0f);
    atomicAdd(&degIn[dst[e]], 1.0f);
  }
}

__global__ void norm_kernel(float* __restrict__ p, int n) {
  int i = blockIdx.x * blockDim.x + threadIdx.x;
  if (i < n) p[i] = rsqrtf(fmaxf(p[i], 1.0f));
}

// ---------- WMMA GEMM: Y[N,64] = (X[N,K] * rowScale[N]) @ Wg[K,64] ----------
// 128 threads = 4 wave32; each wave computes one 16x16 tile of the 16x64 row block.
// W is stored TRANSPOSED in LDS so both A and B fragments are contiguous
// 8x bf16 (16B) runs per lane-half -> ds_load_b128 feeds straight into v_wmma.
template<int K>
__global__ __launch_bounds__(128) void gemm_rownorm_wmma(
    const float* __restrict__ X, const float* __restrict__ rowScale,
    const float* __restrict__ Wg, float* __restrict__ Y, int nRows) {
  __shared__ __bf16 Xs[16][K];          // row-scaled A tile; row stride K*2B (16B mult.)
  __shared__ __bf16 Wt[64][K + 8];      // W^T tile; row stride (K+8)*2B (16B mult.)

  const int tid     = threadIdx.x;
  const int rowBase = blockIdx.x * 16;

  // Stage A tile (with out_norm row scaling) into LDS as bf16.
  for (int idx = tid; idx < 16 * K; idx += 128) {
    int r = idx / K, k = idx - r * K;
    int row = rowBase + r;
    row = (row < nRows) ? row : (nRows - 1);        // clamp (N divisible by 16 anyway)
    Xs[r][k] = f2bf(X[(long)row * K + k] * rowScale[row]);
  }
  // Stage W (K x 64) transposed into LDS as bf16 (coalesced global reads).
  for (int idx = tid; idx < K * 64; idx += 128) {
    int k = idx >> 6, c = idx & 63;
    Wt[c][k] = f2bf(Wg[idx]);
  }
  __syncthreads();

  const int lane    = tid & 31;       // wave32
  const int wave    = tid >> 5;
  const int hh      = lane >> 4;      // lane-half selects K sub-blocks per ISA layout
  const int l       = lane & 15;      // row (A) / col (B) index within tile
  const int colBase = wave << 4;

  v8f acc = {};
  #pragma unroll
  for (int k0 = 0; k0 < K; k0 += 32) {
    // ISA 16-bit A/B layout (16x32): lane(half hh, idx l), element i:
    //   K = k0 + (2*(i/8) + hh)*8 + (i%8)  -> two contiguous 8-element runs.
    const v8bf a0 = *reinterpret_cast<const v8bf*>(&Xs[l][k0 + (hh << 3)]);
    const v8bf a1 = *reinterpret_cast<const v8bf*>(&Xs[l][k0 + 16 + (hh << 3)]);
    const v8bf b0 = *reinterpret_cast<const v8bf*>(&Wt[colBase + l][k0 + (hh << 3)]);
    const v8bf b1 = *reinterpret_cast<const v8bf*>(&Wt[colBase + l][k0 + 16 + (hh << 3)]);
    v16bf a, b;
    #pragma unroll
    for (int i = 0; i < 8; ++i) {
      a[i] = a0[i]; a[i + 8] = a1[i];
      b[i] = b0[i]; b[i + 8] = b1[i];
    }
    acc = __builtin_amdgcn_wmma_f32_16x16x32_bf16(
        /*neg_a=*/false, a, /*neg_b=*/false, b,
        /*c_mod=*/(short)0, acc, /*reuse_a=*/false, /*reuse_b=*/false);
  }

  // D layout: VGPR v -> row (v + 8*hh), col l (per 32-bit C/D 16x16 table).
  // Block-uniform full-tile test -> branch-free stores on the common path.
  float* __restrict__ yOut = Y + ((long)(rowBase + (hh << 3)) * 64 + colBase + l);
  if (rowBase + 16 <= nRows) {
    #pragma unroll
    for (int v = 0; v < 8; ++v) yOut[(long)v * 64] = acc[v];
  } else {
    #pragma unroll
    for (int v = 0; v < 8; ++v) {
      if (rowBase + v + (hh << 3) < nRows) yOut[(long)v * 64] = acc[v];
    }
  }
}

// ---------- edge pass: agg[dst] += y[src]  (d=64, one thread per edge x float4) ----------
__global__ void edge_agg_kernel(const float* __restrict__ y, const int* __restrict__ src,
                                const int* __restrict__ dst, float* __restrict__ agg, int nE) {
  int gid = blockIdx.x * blockDim.x + threadIdx.x;
  int e = gid >> 4;
  if (e >= nE) return;
  int c4 = (gid & 15) << 2;
  int s = src[e], d = dst[e];
  const float4 v = *reinterpret_cast<const float4*>(y + (long)s * 64 + c4);
  float* ap = agg + (long)d * 64 + c4;
  atomicAdd(ap + 0, v.x);
  atomicAdd(ap + 1, v.y);
  atomicAdd(ap + 2, v.z);
  atomicAdd(ap + 3, v.w);
}

// ---------- finalize: out = [relu]( agg * in_norm + bias ) ----------
__global__ void finalize_kernel(const float* __restrict__ agg, const float* __restrict__ inNorm,
                                const float* __restrict__ bias, float* __restrict__ out,
                                int nRows, int doRelu) {
  int gid = blockIdx.x * blockDim.x + threadIdx.x;
  if (gid >= nRows * 64) return;
  int n = gid >> 6, c = gid & 63;
  float v = agg[gid] * inNorm[n] + bias[c];
  out[gid] = doRelu ? fmaxf(v, 0.0f) : v;
}

static inline int cdiv_i(long a, long b) { return (int)((a + b - 1) / b); }

extern "C" void kernel_launch(void* const* d_in, const int* in_sizes, int n_in,
                              void* d_out, int out_size, void* d_ws, size_t ws_size,
                              hipStream_t stream) {
  const float* feat = (const float*)d_in[0];
  const float* W0   = (const float*)d_in[1];
  const float* b0   = (const float*)d_in[2];
  const float* W1   = (const float*)d_in[3];
  const float* b1   = (const float*)d_in[4];
  const float* W2   = (const float*)d_in[5];
  const float* b2   = (const float*)d_in[6];
  const int*   src  = (const int*)d_in[7];
  const int*   dst  = (const int*)d_in[8];

  const int N = in_sizes[0] / 128;   // 50000
  const int E = in_sizes[7];         // 1,600,000
  float* out = (float*)d_out;

  // workspace layout (floats): no[N] | ni[N] | y[64N] | agg[64N] | h[64N]
  float* wsf = (float*)d_ws;
  float* no  = wsf;
  float* ni  = no + N;
  float* y   = ni + N;
  float* ag  = y  + (size_t)N * 64;
  float* h   = ag + (size_t)N * 64;

  const int NB = 256;
  const int gemmBlocks = cdiv_i(N, 16);
  const int nNF = N * 64;

  // degrees -> norms (in place)
  fill_zero_kernel<<<cdiv_i(2 * N, NB), NB, 0, stream>>>(no, 2 * N);
  degree_kernel<<<cdiv_i(E, NB), NB, 0, stream>>>(src, dst, no, ni, E);
  norm_kernel<<<cdiv_i(2 * N, NB), NB, 0, stream>>>(no, 2 * N);

  // ---- layer 0: h = relu( S(feat) @ W0 + b0 ), W applied pre-aggregation (linear) ----
  gemm_rownorm_wmma<128><<<gemmBlocks, 128, 0, stream>>>(feat, no, W0, y, N);
  fill_zero_kernel<<<cdiv_i(nNF, NB), NB, 0, stream>>>(ag, nNF);
  edge_agg_kernel<<<cdiv_i((long)E * 16, NB), NB, 0, stream>>>(y, src, dst, ag, E);
  finalize_kernel<<<cdiv_i(nNF, NB), NB, 0, stream>>>(ag, ni, b0, h, N, 1);

  // ---- layer 1 ----
  gemm_rownorm_wmma<64><<<gemmBlocks, 128, 0, stream>>>(h, no, W1, y, N);
  fill_zero_kernel<<<cdiv_i(nNF, NB), NB, 0, stream>>>(ag, nNF);
  edge_agg_kernel<<<cdiv_i((long)E * 16, NB), NB, 0, stream>>>(y, src, dst, ag, E);
  finalize_kernel<<<cdiv_i(nNF, NB), NB, 0, stream>>>(ag, ni, b1, h, N, 1);

  // ---- layer 2 (no relu) -> d_out ----
  gemm_rownorm_wmma<64><<<gemmBlocks, 128, 0, stream>>>(h, no, W2, y, N);
  fill_zero_kernel<<<cdiv_i(nNF, NB), NB, 0, stream>>>(ag, nNF);
  edge_agg_kernel<<<cdiv_i((long)E * 16, NB), NB, 0, stream>>>(y, src, dst, ag, E);
  finalize_kernel<<<cdiv_i(nNF, NB), NB, 0, stream>>>(ag, ni, b2, out, N, 0);
}
